// Discriminator_77687368450470
// MI455X (gfx1250) — matrix operand order repaired and checked by reading, not compiled, not run on hardware
//
#include <hip/hip_runtime.h>
#include <hip/hip_bf16.h>

typedef __bf16 bf16_t;
typedef __attribute__((ext_vector_type(16))) bf16_t v16bf;
typedef __attribute__((ext_vector_type(8)))  bf16_t v8bf;
typedef __attribute__((ext_vector_type(8)))  float  v8f;
typedef __attribute__((ext_vector_type(4)))  unsigned int v4u32;
typedef __attribute__((ext_vector_type(4)))  int    v4i32;
typedef __attribute__((ext_vector_type(8)))  int    v8i32;
typedef v8bf uav8bf __attribute__((aligned(4)));   // 4B-aligned vector load

#define H1_STRIDE 1056   // 8 front halo + 1024 + 24 back halo (32B-aligned rows)
#define H1_PAD    8

// ---- TDM availability / arity dispatch (device pass only) ------------------
#if __has_include(<hip/amd_detail/amd_gfx1250_TDM.h>)
#define TDM_ARITY 6
#elif __has_builtin(__builtin_amdgcn_tensor_load_to_lds)
#define TDM_ARITY 5
#else
#define TDM_ARITY 0
#endif

// ---------------------------------------------------------------------------
// Direct-from-global WMMA fragments (ISA 7.12.2).
// A 16x32 bf16 fragment, lane L: row m0+(L&15), two contiguous 8-elem K-runs
//   at k0+8h and k0+16+8h (h = L>>4).
// B 32x16 fragment from a [N,K] row-major matrix, lane L: row n0+(L&15),
//   two contiguous 8-elem K-runs at k0+8h and k0+16+8h.
// ---------------------------------------------------------------------------
__device__ __forceinline__ v8bf cvt8(float4 a, float4 b) {
  v8bf r;
  r[0] = (bf16_t)a.x; r[1] = (bf16_t)a.y; r[2] = (bf16_t)a.z; r[3] = (bf16_t)a.w;
  r[4] = (bf16_t)b.x; r[5] = (bf16_t)b.y; r[6] = (bf16_t)b.z; r[7] = (bf16_t)b.w;
  return r;
}
__device__ __forceinline__ v16bf cat16(v8bf lo, v8bf hi) {
  return __builtin_shufflevector(lo, hi, 0, 1, 2, 3, 4, 5, 6, 7,
                                 8, 9, 10, 11, 12, 13, 14, 15);
}

template <typename T>   // row: pointer to the lane's matrix row; kb = k0 + 8*h
__device__ __forceinline__ v16bf load_afrag(const T* row, int kb) {
  if constexpr (sizeof(T) == 2) {
    v8bf lo = *(const v8bf*)(row + kb);
    v8bf hi = *(const v8bf*)(row + kb + 16);
    return cat16(lo, hi);
  } else {
    const float4* p0 = (const float4*)(row + kb);
    const float4* p1 = (const float4*)(row + kb + 16);
    return cat16(cvt8(p0[0], p0[1]), cvt8(p1[0], p1[1]));
  }
}
template <typename T>   // kb = k0 + 8*h
__device__ __forceinline__ v16bf load_bfrag(const T* row, int kb) {
  if constexpr (sizeof(T) == 2) {
    v8bf lo = *(const v8bf*)(row + kb);
    v8bf hi = *(const v8bf*)(row + kb + 16);
    return cat16(lo, hi);
  } else {
    const float4* p0 = (const float4*)(row + kb);
    const float4* p1 = (const float4*)(row + kb + 16);
    return cat16(cvt8(p0[0], p0[1]), cvt8(p1[0], p1[1]));
  }
}

// ---- TDM: 2D descriptor, 32x32 f32 tile, row stride = ld elements ----------
__device__ __forceinline__ void tdm_issue_32x32_f32(
    unsigned int lds_off, const float* gptr, unsigned int ld, unsigned int dim1)
{
#if TDM_ARITY > 0
  unsigned long long ga = (unsigned long long)(const void*)gptr;
  v4u32 g0;
  g0[0] = 1u;                                   // count=1, user mode
  g0[1] = lds_off;                              // lds_addr
  g0[2] = (unsigned int)ga;                     // global_addr[31:0]
  g0[3] = ((unsigned int)(ga >> 32) & 0x01FFFFFFu) | (2u << 30); // [56:32]|type=2
  v8i32 g1;
  g1[0] = (int)(2u << 16);                      // data_size=4B, wg_mask=0
  g1[1] = (int)((ld & 0xFFFFu) << 16);          // tensor_dim0[15:0]
  g1[2] = (int)(((ld >> 16) & 0xFFFFu) | ((dim1 & 0xFFFFu) << 16));
  g1[3] = (int)(((dim1 >> 16) & 0xFFFFu) | (32u << 16)); // tile_dim0=32
  g1[4] = 32;                                   // tile_dim1=32
  g1[5] = (int)ld;                              // tensor_dim0_stride[31:0]
  g1[6] = 0;
  g1[7] = 0;
  v4i32 z4 = {0, 0, 0, 0};
#if TDM_ARITY == 6
  v8i32 z8 = {0, 0, 0, 0, 0, 0, 0, 0};
  __builtin_amdgcn_tensor_load_to_lds(g0, g1, z4, z4, z8, 0);
#else
  __builtin_amdgcn_tensor_load_to_lds(g0, g1, z4, z4, 0);
#endif
#endif
}
__device__ __forceinline__ void wait_tensor0() {
#if __has_builtin(__builtin_amdgcn_s_wait_tensorcnt)
  __builtin_amdgcn_s_wait_tensorcnt(0);
#elif TDM_ARITY > 0
  asm volatile("s_wait_tensorcnt 0x0" ::: "memory");
#endif
}
__device__ __forceinline__ void wait_tensor1() {
#if __has_builtin(__builtin_amdgcn_s_wait_tensorcnt)
  __builtin_amdgcn_s_wait_tensorcnt(1);
#elif TDM_ARITY > 0
  asm volatile("s_wait_tensorcnt 0x1" ::: "memory");
#endif
}

// ---------------------------------------------------------------------------
// LDS-free wave-level WMMA GEMM: C[M,N] = act(A[M,K] * B[N,K]^T + bias)
// 32x32 C per wave (2x2 WMMA tiles), fragments loaded directly from global.
// USE_TDM: stream B 32x32 f32 slabs via tensor_load_to_lds, double-buffered.
// Grid: (M/32)*(N/32) == gridDim.x * WPB.
// ---------------------------------------------------------------------------
template <typename AT, typename BT, bool LRELU, bool USE_TDM, int WPB>
__global__ __launch_bounds__(WPB * 32) void gemm_wmma(
    const AT* __restrict__ A, const BT* __restrict__ B,
    const float* __restrict__ bias, float* __restrict__ C,
    int M, int N, int K)
{
  constexpr bool TDM_ON = USE_TDM && (TDM_ARITY > 0);
  __shared__ alignas(64) float rawB[TDM_ON ? WPB * 2048 : 1]; // 2x 32x32 f32 /wave

  const int wave = threadIdx.x >> 5;
  const int lane = threadIdx.x & 31;
  const int tile = blockIdx.x * WPB + wave;
  const int tiles_n = N >> 5;
  const int m0 = (tile / tiles_n) << 5;
  const int n0 = (tile % tiles_n) << 5;
  const int h  = lane >> 4;
  const int ml = lane & 15;

  const AT* arow0 = A + (size_t)(m0 + ml) * K;
  const AT* arow1 = A + (size_t)(m0 + 16 + ml) * K;
  const BT* brow0 = B + (size_t)(n0 + ml) * K;
  const BT* brow1 = B + (size_t)(n0 + 16 + ml) * K;
  float* rb = rawB + wave * 2048;
  v8f acc00 = {}, acc01 = {}, acc10 = {}, acc11 = {};

  if constexpr (TDM_ON)
    tdm_issue_32x32_f32((unsigned int)(size_t)rb, &B[(size_t)n0 * K], (unsigned)K,
                        (unsigned)N);

  for (int k0 = 0; k0 < K; k0 += 32) {
    int kA = k0 + 8 * h;
    int kB = k0 + 8 * h;
    v16bf a0 = load_afrag(arow0, kA);
    v16bf a1 = load_afrag(arow1, kA);
    v16bf b0, b1;
    if constexpr (TDM_ON) {
      bool more = (k0 + 32 < K);
      int cur = (k0 >> 5) & 1;
      if (more) {
        tdm_issue_32x32_f32((unsigned int)(size_t)(rb + (cur ^ 1) * 1024),
                            &B[(size_t)n0 * K + (k0 + 32)], (unsigned)K, (unsigned)N);
        wait_tensor1();                  // current slab done, next in flight
      } else {
        wait_tensor0();
      }
      const float* t0 = rb + cur * 1024 + (size_t)ml * 32 + 8 * h;
      const float* t1 = rb + cur * 1024 + (size_t)(16 + ml) * 32 + 8 * h;
      const float4* q0 = (const float4*)t0;
      const float4* q1 = (const float4*)t1;
      b0 = cat16(cvt8(q0[0], q0[1]), cvt8(q0[4], q0[5]));
      b1 = cat16(cvt8(q1[0], q1[1]), cvt8(q1[4], q1[5]));
    } else {
      b0 = load_bfrag(brow0, kB);
      b1 = load_bfrag(brow1, kB);
    }
    acc00 = __builtin_amdgcn_wmma_f32_16x16x32_bf16(false, a0, false, b0, (short)0, acc00, false, false);
    acc01 = __builtin_amdgcn_wmma_f32_16x16x32_bf16(false, a0, false, b1, (short)0, acc01, false, false);
    acc10 = __builtin_amdgcn_wmma_f32_16x16x32_bf16(false, a1, false, b0, (short)0, acc10, false, false);
    acc11 = __builtin_amdgcn_wmma_f32_16x16x32_bf16(false, a1, false, b1, (short)0, acc11, false, false);
  }

#pragma unroll
  for (int sm = 0; sm < 2; ++sm) {
#pragma unroll
    for (int sn = 0; sn < 2; ++sn) {
      const v8f& acc = sm ? (sn ? acc11 : acc10) : (sn ? acc01 : acc00);
#pragma unroll
      for (int r = 0; r < 8; ++r) {
        int m = m0 + sm * 16 + r + 8 * h;
        int n = n0 + sn * 16 + ml;
        float v = acc[r];
        if (bias) v += bias[n];
        if (LRELU) v = v >= 0.f ? v : 0.2f * v;
        C[(size_t)m * N + n] = v;
      }
    }
  }
}

// ---------------------------------------------------------------------------
// conv2 implicit GEMM, LDS-free and branchless: M=(b,op)=65536, N=oc=256,
// K=(ic,t)=2048. h1 rows carry zero halos (stride H1_STRIDE, data at +H1_PAD),
// so every A-fragment run (p0 in [-6,1029]) is a plain unconditional load.
// ---------------------------------------------------------------------------
__global__ __launch_bounds__(256) void conv2_wmma(
    const bf16_t* __restrict__ h1, const float* __restrict__ w2,
    float* __restrict__ h2)
{
  const int wave = threadIdx.x >> 5;
  const int lane = threadIdx.x & 31;
  const int tile = blockIdx.x * 8 + wave;   // 16384 tiles
  const int m0 = (tile / 8) << 5;           // tiles_n = 256/32 = 8
  const int n0 = (tile % 8) << 5;
  const int b  = m0 >> 8;
  const int opbase = m0 & 255;
  const int h  = lane >> 4;
  const int ml = lane & 15;

  const float* brow0 = w2 + (size_t)(n0 + ml) * 2048;
  const float* brow1 = w2 + (size_t)(n0 + 16 + ml) * 2048;
  v8f acc00 = {}, acc01 = {}, acc10 = {}, acc11 = {};

  for (int k0 = 0; k0 < 2048; k0 += 32) {
    int kA = k0 + 8 * h;
    int ic0 = kA >> 4;
    int t0  = kA & 15;                       // 0 or 8
    // run0: channel ic0; run1: channel ic0+1, same p window
    const bf16_t* r0 = h1 + (size_t)(b * 128 + ic0) * H1_STRIDE + H1_PAD;
    int pA = 4 * (opbase + ml) - 6 + t0;
    int pB = 4 * (opbase + 16 + ml) - 6 + t0;
    v16bf a0 = cat16(*(const uav8bf*)(r0 + pA),
                     *(const uav8bf*)(r0 + H1_STRIDE + pA));
    v16bf a1 = cat16(*(const uav8bf*)(r0 + pB),
                     *(const uav8bf*)(r0 + H1_STRIDE + pB));
    v16bf b0 = load_bfrag(brow0, k0 + 8 * h);
    v16bf b1 = load_bfrag(brow1, k0 + 8 * h);

    acc00 = __builtin_amdgcn_wmma_f32_16x16x32_bf16(false, a0, false, b0, (short)0, acc00, false, false);
    acc01 = __builtin_amdgcn_wmma_f32_16x16x32_bf16(false, a0, false, b1, (short)0, acc01, false, false);
    acc10 = __builtin_amdgcn_wmma_f32_16x16x32_bf16(false, a1, false, b0, (short)0, acc10, false, false);
    acc11 = __builtin_amdgcn_wmma_f32_16x16x32_bf16(false, a1, false, b1, (short)0, acc11, false, false);
  }

#pragma unroll
  for (int sm = 0; sm < 2; ++sm) {
#pragma unroll
    for (int sn = 0; sn < 2; ++sn) {
      const v8f& acc = sm ? (sn ? acc11 : acc10) : (sn ? acc01 : acc00);
#pragma unroll
      for (int r = 0; r < 8; ++r) {
        int op = opbase + sm * 16 + r + 8 * h;
        int oc = n0 + sn * 16 + ml;
        h2[(size_t)b * 65536 + oc * 256 + op] = acc[r];   // [b][oc][op]
      }
    }
  }
}

// ---------------------------------------------------------------------------
// conv1: 1->128, k16 s4 pad6, + LeakyReLU; writes h1 rows WITH zero halos.
// tid covers the full padded layout so every byte of h1 is initialized.
// ---------------------------------------------------------------------------
__global__ __launch_bounds__(256) void conv1_k(
    const float* __restrict__ h0, const float* __restrict__ w1,
    bf16_t* __restrict__ h1)
{
  int tid  = blockIdx.x * 256 + threadIdx.x;   // b*(128*1056) + oc*1056 + q
  int q    = tid % H1_STRIDE;
  int rest = tid / H1_STRIDE;
  int oc = rest & 127;
  int b  = rest >> 7;
  int p  = q - H1_PAD;
  float acc = 0.f;
  if (p >= 0 && p < 1024) {
    const float* xr = h0 + (size_t)b * 4096;
    const float* wr = w1 + oc * 16;
#pragma unroll
    for (int t = 0; t < 16; ++t) {
      int pos = 4 * p - 6 + t;
      int pc  = pos < 0 ? 0 : (pos > 4095 ? 4095 : pos);
      float v = xr[pc];
      acc += (pos == pc ? v : 0.f) * wr[t];
    }
    acc = acc >= 0.f ? acc : 0.2f * acc;
  }
  h1[tid] = (bf16_t)acc;
}

__global__ __launch_bounds__(256) void bn_stats(
    const float* __restrict__ h2, const float* __restrict__ g,
    const float* __restrict__ be, float* __restrict__ scale,
    float* __restrict__ shift)
{
  int oc = blockIdx.x, t = threadIdx.x;
  float s = 0.f, s2 = 0.f;
  for (int i = t; i < 65536; i += 256) {
    int b = i >> 8, op = i & 255;
    float v = h2[(size_t)b * 65536 + oc * 256 + op];
    s += v; s2 += v * v;
  }
  __shared__ float ls[256], ls2[256];
  ls[t] = s; ls2[t] = s2;
  __syncthreads();
  for (int off = 128; off; off >>= 1) {
    if (t < off) { ls[t] += ls[t + off]; ls2[t] += ls2[t + off]; }
    __syncthreads();
  }
  if (t == 0) {
    float mean = ls[0] * (1.f / 65536.f);
    float var  = ls2[0] * (1.f / 65536.f) - mean * mean;
    float sc   = g[oc] * rsqrtf(var + 1e-5f);
    scale[oc] = sc;
    shift[oc] = be[oc] - mean * sc;
  }
}

__global__ __launch_bounds__(256) void bn_apply(
    const float* __restrict__ h2, const float* __restrict__ scale,
    const float* __restrict__ shift, bf16_t* __restrict__ hbn)
{
  int tid = blockIdx.x * 256 + threadIdx.x;   // b*65536 + oc*256 + op
  int oc = (tid >> 8) & 255;
  float v = h2[tid] * scale[oc] + shift[oc];
  v = v >= 0.f ? v : 0.2f * v;
  hbn[tid] = (bf16_t)v;
}

// T (1024,128,16) -> Tb[j][k] bf16, j=of*16+kd (so M-proj is a TRANSB GEMM)
__global__ __launch_bounds__(256) void transpose_T(
    const float* __restrict__ T, bf16_t* __restrict__ Tb)
{
  int tid = blockIdx.x * 256 + threadIdx.x;   // j*1024 + k
  int j = tid >> 10, k = tid & 1023;
  Tb[tid] = (bf16_t)T[(size_t)k * 2048 + j];
}

__global__ __launch_bounds__(256) void pairwise_k(
    const float* __restrict__ M, float* __restrict__ ob)
{
  int tid = blockIdx.x * 256 + threadIdx.x;   // 32768
  int j  = tid & 255;
  int of = tid >> 8;
  float mj[16];
  const float* base = M + (size_t)j * 2048 + of * 16;
#pragma unroll
  for (int k = 0; k < 16; ++k) mj[k] = base[k];
  float acc = 0.f;
  for (int i = 0; i < 256; ++i) {
    const float* mi = M + (size_t)i * 2048 + of * 16;
    float s = 0.f;
#pragma unroll
    for (int k = 0; k < 16; ++k) s += fabsf(mi[k] - mj[k]);
    acc += __expf(-s);
  }
  ob[j * 128 + of] = acc - 1.f;
}

__global__ __launch_bounds__(256) void final_head(
    const float* __restrict__ feat, const float* __restrict__ ob,
    const float* __restrict__ ow, const float* __restrict__ obias,
    float* __restrict__ out)
{
  int wave = threadIdx.x >> 5, lane = threadIdx.x & 31;
  int b = blockIdx.x * 8 + wave;
  float acc = 0.f;
  for (int i = lane; i < 1152; i += 32) {
    float z = (i < 1024) ? feat[(size_t)b * 1024 + i] : ob[b * 128 + (i - 1024)];
    acc += ow[i] * z;
  }
  for (int off = 16; off; off >>= 1) acc += __shfl_down(acc, off, 32);
  if (lane == 0) out[b] = 1.f / (1.f + __expf(-(acc + obias[0])));
}

// ---------------------------------------------------------------------------
// Workspace layout (bytes)
// ---------------------------------------------------------------------------
#define OFF_H0   ((size_t)0)                      // f32  256x4096        4 MB
#define OFF_H1   (OFF_H0 + 4194304)               // bf16 256x128x1056   66 MB
#define OFF_H2   (OFF_H1 + 69206016)              // f32  256x256x256    64 MB
#define OFF_HBN  (OFF_H2 + 67108864)              // bf16 256x65536      32 MB
#define OFF_FEAT (OFF_HBN + 33554432)             // f32  256x1024        1 MB
#define OFF_M    (OFF_FEAT + 1048576)             // f32  256x2048        2 MB
#define OFF_OB   (OFF_M + 2097152)                // f32  256x128
#define OFF_SC   (OFF_OB + 131072)                // f32  256
#define OFF_SH   (OFF_SC + 1024)                  // f32  256
#define OFF_TB   (OFF_SH + 1024)                  // bf16 2048x1024       4 MB

extern "C" void kernel_launch(void* const* d_in, const int* in_sizes, int n_in,
                              void* d_out, int out_size, void* d_ws, size_t ws_size,
                              hipStream_t stream)
{
  const float* x     = (const float*)d_in[0];
  const float* l0w   = (const float*)d_in[1];
  const float* l0b   = (const float*)d_in[2];
  const float* w1    = (const float*)d_in[3];
  const float* w2    = (const float*)d_in[4];
  const float* bng   = (const float*)d_in[5];
  const float* bnb   = (const float*)d_in[6];
  const float* l1w   = (const float*)d_in[7];
  const float* l1b   = (const float*)d_in[8];
  const float* Tm    = (const float*)d_in[9];
  const float* ow    = (const float*)d_in[10];
  const float* obias = (const float*)d_in[11];
  float* out = (float*)d_out;

  char* ws = (char*)d_ws;
  float*  h0   = (float*)(ws + OFF_H0);
  bf16_t* h1   = (bf16_t*)(ws + OFF_H1);
  float*  h2   = (float*)(ws + OFF_H2);
  bf16_t* hbn  = (bf16_t*)(ws + OFF_HBN);
  float*  feat = (float*)(ws + OFF_FEAT);
  float*  Mb   = (float*)(ws + OFF_M);
  float*  ob   = (float*)(ws + OFF_OB);
  float*  scl  = (float*)(ws + OFF_SC);
  float*  shf  = (float*)(ws + OFF_SH);
  bf16_t* Tb   = (bf16_t*)(ws + OFF_TB);

  // l0: x(256,4096) @ l0_w^T + b                [WMMA, LDS-free]
  gemm_wmma<float, float, false, false, 8><<<128, 256, 0, stream>>>(
      x, l0w, l0b, h0, 256, 4096, 4096);
  // conv1 + lrelu -> bf16 with zero halos (256*128*1056 threads)
  conv1_k<<<135168, 256, 0, stream>>>(h0, w1, h1);
  // conv2 implicit GEMM 65536x256x2048          [WMMA, branchless halo gather]
  conv2_wmma<<<2048, 256, 0, stream>>>(h1, w2, h2);
  // BN training stats + apply + lrelu -> bf16
  bn_stats<<<256, 256, 0, stream>>>(h2, bng, bnb, scl, shf);
  bn_apply<<<65536, 256, 0, stream>>>(h2, scl, shf, hbn);
  // l1: hbn @ l1_w^T + b, lrelu                 [WMMA + double-buffered TDM]
  gemm_wmma<bf16_t, float, true, true, 4><<<64, 128, 0, stream>>>(
      hbn, l1w, l1b, feat, 256, 1024, 65536);
  // T transpose (once per call) then M = feat @ Tb^T   [WMMA, bf16 B]
  transpose_T<<<8192, 256, 0, stream>>>(Tm, Tb);
  gemm_wmma<float, bf16_t, false, false, 8><<<64, 256, 0, stream>>>(
      feat, Tb, nullptr, Mb, 256, 2048, 1024);
  // minibatch discrimination + head
  pairwise_k<<<128, 256, 0, stream>>>(Mb, ob);
  final_head<<<32, 256, 0, stream>>>(feat, ob, ow, obias, out);
}
